// myModel_50792283242756
// MI455X (gfx1250) — matrix operand order repaired and checked by reading, not compiled, not run on hardware
//
#include <hip/hip_runtime.h>

// ---------------------------------------------------------------------------
// CDNA5 (gfx1250) GNN message-passing kernel set, round 4.
//  - bf16 state mirrors: random gathers + h-path A fragments load bf16.
//  - Weights staged in LDS per block via the Tensor Data Mover
//    (tensor_load_to_lds + s_wait_tensorcnt), cooperative-copy fallback.
//  - B fragments from ds_load_b128; GRU A fragments hoisted.
// ---------------------------------------------------------------------------

typedef __attribute__((ext_vector_type(16))) __bf16          bf16x16;
typedef __attribute__((ext_vector_type(8)))  float           f32x8;
typedef __attribute__((ext_vector_type(16))) unsigned short  u16x16;
typedef __attribute__((ext_vector_type(4)))  unsigned int    u32x4;
typedef __attribute__((ext_vector_type(4)))  float           f32x4;
typedef __attribute__((ext_vector_type(8)))  int             i32x8;
typedef __attribute__((ext_vector_type(4)))  int             i32x4;

union FragAB {
    bf16x16 bf;
    u16x16  us;
    u32x4   q[2];
};

__device__ __forceinline__ unsigned short f32_to_bf16(float f) {
    unsigned int u = __float_as_uint(f);
    unsigned int lsb = (u >> 16) & 1u;
    u += 0x7fffu + lsb;                      // round-to-nearest-even
    return (unsigned short)(u >> 16);
}

__device__ __forceinline__ float selu_f(float x) {
    const float kScale = 1.0507009873554805f;
    const float kAlpha = 1.6732632423543772f;
    return x > 0.f ? kScale * x : kScale * kAlpha * (expf(x) - 1.f);
}

__device__ __forceinline__ float sigmoid_f(float x) {
    return 1.f / (1.f + expf(-x));
}

__device__ __forceinline__ f32x8 wmma_bf16(bf16x16 a, bf16x16 b, f32x8 c) {
    // D = A(16x32 bf16) x B(32x16 bf16) + C(16x16 f32)
    return __builtin_amdgcn_wmma_f32_16x16x32_bf16(
        false, a, false, b, (short)0, c, false, false);
}

// ---------------------------------------------------------------------------
// Tensor Data Mover: 1-D contiguous tile of `ndwords` DWORDs global -> LDS.
// D# per ISA 8.3/8.4: group0 {count=1, lds_addr, global_addr, type=2},
// group1 {data_size=4B, tensor_dim0 = tile_dim0 = stride = ndwords, dim1=1}.
// This toolchain exposes the 6-arg builtin:
//   (uint32x4 g0, int32x8 g1, int32x4, int32x4, int32x8, i32 cpol)
// ---------------------------------------------------------------------------
#if __has_builtin(__builtin_amdgcn_tensor_load_to_lds)
#define HAVE_TDM 1
__device__ __forceinline__ void tdm_load_to_lds(unsigned lds_addr,
                                                const void* gsrc,
                                                unsigned ndwords) {
    unsigned long long ga = (unsigned long long)(size_t)gsrc;
    u32x4 g0;
    g0[0] = 1u;                                                  // count=1 (valid user D#)
    g0[1] = lds_addr;                                            // LDS byte address
    g0[2] = (unsigned)(ga & 0xffffffffu);                        // global_addr[31:0]
    g0[3] = (unsigned)((ga >> 32) & 0x01ffffffu) | (2u << 30);   // addr[56:32] | type=2
    i32x8 g1;
    g1[0] = (int)(2u << 16);                                     // data_size = 4 bytes
    g1[1] = (int)((ndwords & 0xffffu) << 16);                    // tensor_dim0[15:0]
    g1[2] = (int)(((ndwords >> 16) & 0xffffu) | (1u << 16));     // dim0 hi | tensor_dim1=1
    g1[3] = (int)((ndwords & 0xffffu) << 16);                    // tile_dim0 = ndwords
    g1[4] = 0;                                                   // tile_dim1/2 unused
    g1[5] = (int)ndwords;                                        // tensor_dim0_stride
    g1[6] = 0;
    g1[7] = 0;
    i32x4 gz4;
    gz4[0] = gz4[1] = gz4[2] = gz4[3] = 0;
    i32x8 gz8;
#pragma unroll
    for (int i = 0; i < 8; ++i) gz8[i] = 0;
    __builtin_amdgcn_tensor_load_to_lds(g0, g1, gz4, gz4, gz8, 0);
}
#else
#define HAVE_TDM 0
#endif

// Stage `ndwords` DWORDs of weights into LDS (TDM if available, else copy).
// Must be called by ALL threads of the block (contains no barrier itself).
__device__ __forceinline__ void stage_weights(unsigned short* sdst,
                                              const unsigned short* gsrc,
                                              int ndwords) {
#if HAVE_TDM
    if (threadIdx.x < 32) {
        tdm_load_to_lds((unsigned)(size_t)&sdst[0], gsrc, (unsigned)ndwords);
        __builtin_amdgcn_s_wait_tensorcnt(0);
    }
#else
    const u32x4* g = reinterpret_cast<const u32x4*>(gsrc);
    u32x4* s = reinterpret_cast<u32x4*>(sdst);
    for (int i = threadIdx.x; i < ndwords / 4; i += 256) s[i] = g[i];
#endif
}

// A fragment (16x32, M = lane&15) from an fp32 row pointer (converted).
// ISA layout: lanes<16 -> K = 0..7,16..23 ; lanes>=16 -> +8.
__device__ __forceinline__ bf16x16 load_a_frag_f32(const float* __restrict__ p, int lane) {
    const int klo = (lane >= 16) ? 8 : 0;
    f32x4 a0 = *reinterpret_cast<const f32x4*>(p + klo);
    f32x4 a1 = *reinterpret_cast<const f32x4*>(p + klo + 4);
    f32x4 b0 = *reinterpret_cast<const f32x4*>(p + klo + 16);
    f32x4 b1 = *reinterpret_cast<const f32x4*>(p + klo + 20);
    FragAB f;
#pragma unroll
    for (int i = 0; i < 4; ++i) {
        f.us[i]      = f32_to_bf16(a0[i]);
        f.us[4 + i]  = f32_to_bf16(a1[i]);
        f.us[8 + i]  = f32_to_bf16(b0[i]);
        f.us[12 + i] = f32_to_bf16(b1[i]);
    }
    return f.bf;
}

// A fragment directly from a bf16 row pointer (no conversion): two 16B loads.
__device__ __forceinline__ bf16x16 load_a_frag_bf16(const unsigned short* __restrict__ p,
                                                    int lane) {
    const int klo = (lane >= 16) ? 8 : 0;
    FragAB f;
    f.q[0] = *reinterpret_cast<const u32x4*>(p + klo);        // K klo..klo+7
    f.q[1] = *reinterpret_cast<const u32x4*>(p + klo + 16);   // K klo+16..klo+23
    return f.bf;
}

// B fragment (32x16, N = lane&15) from [N_total][ldK] bf16 weights
// (LDS); each lane reads K klo..klo+7 and klo+16..klo+23 (klo = 0 or 8).
__device__ __forceinline__ bf16x16 load_b_frag(const unsigned short* Wt,
                                               int ldK, int n0, int kbase, int lane) {
    const unsigned short* p = Wt + (size_t)(n0 + (lane & 15)) * ldK
                                 + kbase + ((lane >= 16) ? 8 : 0);
    FragAB f;
    f.q[0] = *reinterpret_cast<const u32x4*>(p);
    f.q[1] = *reinterpret_cast<const u32x4*>(p + 16);
    return f.bf;
}

__device__ __forceinline__ f32x8 zero_f32x8() {
    f32x8 z;
#pragma unroll
    for (int i = 0; i < 8; ++i) z[i] = 0.f;
    return z;
}

// ---------------------------------------------------------------------------
// Message kernel: msg = selu(concat(state[first], state[second]) @ W + b),
// agg[second] += msg (atomic scatter).  A: [16 rows, K=128] bf16 gathers,
// B: LDS-staged weights.  One wave per 16-row tile, 8 tiles per block.
// ---------------------------------------------------------------------------
__global__ void msg_kernel(const unsigned short* __restrict__ state_bf, // [E][64] bf16
                           const int* __restrict__ first,
                           const int* __restrict__ second,
                           const unsigned short* __restrict__ Wt,       // [64][128] bf16
                           const float* __restrict__ bias,              // [64]
                           float* __restrict__ agg,                     // [E][64]
                           int M) {
    __shared__ unsigned short sW[64 * 128];                  // 16 KB
    stage_weights(sW, Wt, 64 * 128 / 2);                     // 4096 DWORDs
    __syncthreads();

    const int lane = threadIdx.x & 31;
    const int wave = threadIdx.x >> 5;
    const int m0 = (blockIdx.x * (blockDim.x >> 5) + wave) * 16;
    if (m0 >= M) return;

    int aRow = m0 + (lane & 15);
    if (aRow >= M) aRow = M - 1;
    const int fIdx = first[aRow];
    const int sIdx = second[aRow];

    f32x8 acc[4];
#pragma unroll
    for (int nt = 0; nt < 4; ++nt) acc[nt] = zero_f32x8();

#pragma unroll
    for (int ks = 0; ks < 4; ++ks) {
        const unsigned short* base = (ks < 2)
            ? (state_bf + (size_t)fIdx * 64 + ks * 32)           // main half  (K 0..63)
            : (state_bf + (size_t)sIdx * 64 + (ks - 2) * 32);    // neigh half (K 64..127)
        bf16x16 a = load_a_frag_bf16(base, lane);
#pragma unroll
        for (int nt = 0; nt < 4; ++nt) {
            bf16x16 b = load_b_frag(sW, 128, nt * 16, ks * 32, lane);
            acc[nt] = wmma_bf16(a, b, acc[nt]);
        }
    }

    const int col0   = lane & 15;
    const int rbase0 = (lane >= 16) ? 8 : 0;
    int dsts[8];
#pragma unroll
    for (int i = 0; i < 8; ++i) {
        const int m = m0 + rbase0 + i;
        dsts[i] = (m < M) ? second[m] : -1;
    }
#pragma unroll
    for (int nt = 0; nt < 4; ++nt) {
        const int col = nt * 16 + col0;
        const float bb = bias[col];
#pragma unroll
        for (int i = 0; i < 8; ++i) {
            if (dsts[i] >= 0) {
                float v = selu_f(acc[nt][i] + bb);
                atomicAdd(agg + (size_t)dsts[i] * 64 + col, v);
            }
        }
    }
}

// ---------------------------------------------------------------------------
// GRU kernel (Keras reset_after=True):
//   xm = agg @ Wx + bx ; hm = h @ Wh + bh   ([16,192], K=64)
//   z = sig(xz+hz); r = sig(xr+hr); c = tanh(xh + r*hh); h' = z*h + (1-z)*c
// f32 state updated in place (row-local); bf16 mirror ping-pongs in->out.
// A fragments hoisted; weights staged in LDS via TDM.
// ---------------------------------------------------------------------------
__global__ void gru_kernel(const float* __restrict__ agg,                // [E][64]
                           float* __restrict__ hf,                       // [E][64] f32 in/out
                           const unsigned short* __restrict__ hbf_in,    // [E][64] bf16
                           unsigned short* __restrict__ hbf_out,         // [E][64] bf16
                           const unsigned short* __restrict__ Wx_t,      // [192][64] bf16
                           const unsigned short* __restrict__ Wh_t,      // [192][64] bf16
                           const float* __restrict__ bx,                 // [192]
                           const float* __restrict__ bh,                 // [192]
                           int E) {
    __shared__ unsigned short sWx[192 * 64];                 // 24 KB
    __shared__ unsigned short sWh[192 * 64];                 // 24 KB
    stage_weights(sWx, Wx_t, 192 * 64 / 2);                  // 6144 DWORDs
    stage_weights(sWh, Wh_t, 192 * 64 / 2);
    __syncthreads();

    const int lane = threadIdx.x & 31;
    const int wave = threadIdx.x >> 5;
    const int m0 = (blockIdx.x * (blockDim.x >> 5) + wave) * 16;
    if (m0 >= E) return;

    int aRow = m0 + (lane & 15);
    if (aRow >= E) aRow = E - 1;

    // Hoisted A fragments: x-path (f32 agg, converted) + h-path (bf16 mirror).
    bf16x16 ax0 = load_a_frag_f32 (agg    + (size_t)aRow * 64,      lane);
    bf16x16 ax1 = load_a_frag_f32 (agg    + (size_t)aRow * 64 + 32, lane);
    bf16x16 ah0 = load_a_frag_bf16(hbf_in + (size_t)aRow * 64,      lane);
    bf16x16 ah1 = load_a_frag_bf16(hbf_in + (size_t)aRow * 64 + 32, lane);

    const int col0   = lane & 15;
    const int rbase0 = (lane >= 16) ? 8 : 0;

#pragma unroll
    for (int nt = 0; nt < 4; ++nt) {            // 16-wide column tile of the state
        f32x8 xz = zero_f32x8(), xr = zero_f32x8(), xh = zero_f32x8();
        f32x8 hz = zero_f32x8(), hr = zero_f32x8(), hh = zero_f32x8();
#pragma unroll
        for (int ks = 0; ks < 2; ++ks) {        // K = 64 -> 2 steps of 32
            bf16x16 a_x = ks ? ax1 : ax0;
            bf16x16 a_h = ks ? ah1 : ah0;
            xz = wmma_bf16(a_x, load_b_frag(sWx, 64,       nt * 16, ks * 32, lane), xz);
            xr = wmma_bf16(a_x, load_b_frag(sWx, 64,  64 + nt * 16, ks * 32, lane), xr);
            xh = wmma_bf16(a_x, load_b_frag(sWx, 64, 128 + nt * 16, ks * 32, lane), xh);
            hz = wmma_bf16(a_h, load_b_frag(sWh, 64,       nt * 16, ks * 32, lane), hz);
            hr = wmma_bf16(a_h, load_b_frag(sWh, 64,  64 + nt * 16, ks * 32, lane), hr);
            hh = wmma_bf16(a_h, load_b_frag(sWh, 64, 128 + nt * 16, ks * 32, lane), hh);
        }

        const int col = nt * 16 + col0;
        const float bxz = bx[col],       bhz = bh[col];
        const float bxr = bx[64 + col],  bhr = bh[64 + col];
        const float bxh = bx[128 + col], bhc = bh[128 + col];
#pragma unroll
        for (int i = 0; i < 8; ++i) {
            const int row = m0 + rbase0 + i;
            if (row < E) {
                float z = sigmoid_f(xz[i] + bxz + hz[i] + bhz);
                float r = sigmoid_f(xr[i] + bxr + hr[i] + bhr);
                float c = tanhf(xh[i] + bxh + r * (hh[i] + bhc));
                float hold = hf[(size_t)row * 64 + col];
                float hnew = z * hold + (1.f - z) * c;
                hf[(size_t)row * 64 + col] = hnew;
                hbf_out[(size_t)row * 64 + col] = f32_to_bf16(hnew);
            }
        }
    }
}

// ---------------------------------------------------------------------------
// Prep kernels: state copy/pad (f32 + bf16 mirror) + bf16 transposed weights.
// ---------------------------------------------------------------------------
__global__ void prep_states(const float* __restrict__ gs,   // [E][64]
                            const float* __restrict__ ns,   // [E][59]
                            float* __restrict__ link_f, unsigned short* __restrict__ link_bf,
                            float* __restrict__ node_f, unsigned short* __restrict__ node_bf,
                            int E) {
    const int idx = blockIdx.x * blockDim.x + threadIdx.x;
    if (idx >= E * 64) return;
    const int e = idx >> 6, c = idx & 63;
    const float lv = gs[idx];
    const float nv = (c < 59) ? ns[(size_t)e * 59 + c] : 0.f;
    link_f[idx] = lv;
    link_bf[idx] = f32_to_bf16(lv);
    node_f[idx] = nv;
    node_bf[idx] = f32_to_bf16(nv);
}

// dst[n*K + k] = bf16(src[k*N + n])   (generic transpose -> [N][K] bf16)
__global__ void transpose_bf16(const float* __restrict__ src,
                               unsigned short* __restrict__ dst, int K, int N) {
    const int idx = blockIdx.x * blockDim.x + threadIdx.x;
    if (idx >= K * N) return;
    const int n = idx / K, k = idx % K;
    dst[idx] = f32_to_bf16(src[k * N + n]);
}

// Node message weights: W_nmsg [118][59] -> [64][128] bf16, padded so that
// concat([node_pad[first](64) | node_pad[second](64)]) matches K = 0..127.
__global__ void prep_wnmsg(const float* __restrict__ W, unsigned short* __restrict__ dst) {
    const int idx = blockIdx.x * blockDim.x + threadIdx.x;
    if (idx >= 64 * 128) return;
    const int n = idx >> 7, k = idx & 127;
    float v = 0.f;
    if (n < 59) {
        if (k < 59)                  v = W[k * 59 + n];
        else if (k >= 64 && k < 123) v = W[(k - 5) * 59 + n];
    }
    dst[idx] = f32_to_bf16(v);
}

// Node GRU weights: W [59][177] -> [192][64] bf16; padded col g*64+j <- g*59+j.
__global__ void prep_wgru_n(const float* __restrict__ W, unsigned short* __restrict__ dst) {
    const int idx = blockIdx.x * blockDim.x + threadIdx.x;
    if (idx >= 192 * 64) return;
    const int n = idx >> 6, k = idx & 63;
    const int g = n >> 6, j = n & 63;
    float v = 0.f;
    if (j < 59 && k < 59) v = W[k * 177 + g * 59 + j];
    dst[idx] = f32_to_bf16(v);
}

__global__ void prep_bias_n(const float* __restrict__ bn,   // [59]
                            const float* __restrict__ bxn,  // [177]
                            const float* __restrict__ bhn,  // [177]
                            float* __restrict__ bnm_p,      // [64]
                            float* __restrict__ bxn_p,      // [192]
                            float* __restrict__ bhn_p) {    // [192]
    const int i = threadIdx.x;              // 192 threads
    if (i < 64) bnm_p[i] = (i < 59) ? bn[i] : 0.f;
    const int g = i >> 6, j = i & 63;
    float vx = 0.f, vh = 0.f;
    if (j < 59) { vx = bxn[g * 59 + j]; vh = bhn[g * 59 + j]; }
    bxn_p[i] = vx;
    bhn_p[i] = vh;
}

// ---------------------------------------------------------------------------
// Readout: pooled[g] = sum over edges of [link(64) | node(59)], then MLP.
// ---------------------------------------------------------------------------
__global__ void pool_kernel(const float* __restrict__ link,
                            const float* __restrict__ node,
                            const int* __restrict__ gids,
                            float* __restrict__ pooled,     // [G][128] (123 used)
                            int E) {
    const long long idx = (long long)blockIdx.x * blockDim.x + threadIdx.x;
    const long long e = idx >> 7;
    const int c = (int)(idx & 127);
    if (e >= E) return;
    float v;
    if (c < 64)        v = link[e * 64 + c];
    else if (c < 123)  v = node[e * 64 + (c - 64)];
    else return;
    atomicAdd(pooled + (size_t)gids[e] * 128 + c, v);
}

__global__ void mlp1_kernel(const float* __restrict__ pooled, const float* __restrict__ W1,
                            const float* __restrict__ b1, float* __restrict__ h1) {
    const int g = blockIdx.x, r = threadIdx.x;   // 256 threads
    float s = b1[r];
    for (int k = 0; k < 123; ++k) s += pooled[(size_t)g * 128 + k] * W1[k * 256 + r];
    h1[(size_t)g * 256 + r] = selu_f(s);
}

__global__ void mlp2_kernel(const float* __restrict__ h1, const float* __restrict__ W2,
                            const float* __restrict__ b2, float* __restrict__ h2) {
    const int g = blockIdx.x, r = threadIdx.x;
    float s = b2[r];
    for (int k = 0; k < 256; ++k) s += h1[(size_t)g * 256 + k] * W2[k * 256 + r];
    h2[(size_t)g * 256 + r] = selu_f(s);
}

__global__ void mlp3_kernel(const float* __restrict__ h2, const float* __restrict__ W3,
                            const float* __restrict__ b3, float* __restrict__ out, int G) {
    const int g = blockIdx.x * blockDim.x + threadIdx.x;
    if (g >= G) return;
    float s = b3[0];
    for (int k = 0; k < 256; ++k) s += h2[(size_t)g * 256 + k] * W3[k];
    out[g] = s;
}

// ---------------------------------------------------------------------------
// Host launcher
// ---------------------------------------------------------------------------
extern "C" void kernel_launch(void* const* d_in, const int* in_sizes, int n_in,
                              void* d_out, int out_size, void* d_ws, size_t ws_size,
                              hipStream_t stream) {
    const float* graph_state   = (const float*)d_in[0];
    const float* node_state    = (const float*)d_in[1];
    const int*   states_first  = (const int*)d_in[2];
    const int*   states_second = (const int*)d_in[3];
    const int*   states_gids   = (const int*)d_in[4];
    const float* W_msg  = (const float*)d_in[6];
    const float* b_msg  = (const float*)d_in[7];
    const float* Wx_e   = (const float*)d_in[8];
    const float* Wh_e   = (const float*)d_in[9];
    const float* bx_e   = (const float*)d_in[10];
    const float* bh_e   = (const float*)d_in[11];
    const float* W_nmsg = (const float*)d_in[12];
    const float* b_nmsg = (const float*)d_in[13];
    const float* Wx_n   = (const float*)d_in[14];
    const float* Wh_n   = (const float*)d_in[15];
    const float* bx_n   = (const float*)d_in[16];
    const float* bh_n   = (const float*)d_in[17];
    const float* W1 = (const float*)d_in[18];
    const float* b1 = (const float*)d_in[19];
    const float* W2 = (const float*)d_in[20];
    const float* b2 = (const float*)d_in[21];
    const float* W3 = (const float*)d_in[22];
    const float* b3 = (const float*)d_in[23];
    (void)n_in; (void)ws_size;

    const int M = in_sizes[2];       // adjacency pairs
    const int E = in_sizes[4];       // edge states
    const int G = out_size;          // graphs

    char* ws = (char*)d_ws;
    size_t off = 0;
    auto carve = [&](size_t bytes) -> void* {
        off = (off + 255) & ~(size_t)255;
        void* p = ws + off;
        off += bytes;
        return p;
    };

    float*          link_f  = (float*)carve((size_t)E * 64 * 4);
    float*          node_f  = (float*)carve((size_t)E * 64 * 4);
    unsigned short* link_bf0 = (unsigned short*)carve((size_t)E * 64 * 2);
    unsigned short* link_bf1 = (unsigned short*)carve((size_t)E * 64 * 2);
    unsigned short* node_bf0 = (unsigned short*)carve((size_t)E * 64 * 2);
    unsigned short* node_bf1 = (unsigned short*)carve((size_t)E * 64 * 2);
    float*          agg     = (float*)carve((size_t)E * 64 * 4);
    unsigned short* Wm_t  = (unsigned short*)carve(64 * 128 * 2);
    unsigned short* Wnm_t = (unsigned short*)carve(64 * 128 * 2);
    unsigned short* Wxe_t = (unsigned short*)carve(192 * 64 * 2);
    unsigned short* Whe_t = (unsigned short*)carve(192 * 64 * 2);
    unsigned short* Wxn_t = (unsigned short*)carve(192 * 64 * 2);
    unsigned short* Whn_t = (unsigned short*)carve(192 * 64 * 2);
    float* bnm_p  = (float*)carve(64 * 4);
    float* bxn_p  = (float*)carve(192 * 4);
    float* bhn_p  = (float*)carve(192 * 4);
    float* pooled = (float*)carve((size_t)G * 128 * 4);
    float* h1     = (float*)carve((size_t)G * 256 * 4);
    float* h2     = (float*)carve((size_t)G * 256 * 4);

    // ---- prep: state copy/pad + bf16 transposed weights -------------------
    prep_states<<<(E * 64 + 255) / 256, 256, 0, stream>>>(
        graph_state, node_state, link_f, link_bf0, node_f, node_bf0, E);
    transpose_bf16<<<(128 * 64 + 255) / 256, 256, 0, stream>>>(W_msg, Wm_t, 128, 64);
    transpose_bf16<<<(64 * 192 + 255) / 256, 256, 0, stream>>>(Wx_e, Wxe_t, 64, 192);
    transpose_bf16<<<(64 * 192 + 255) / 256, 256, 0, stream>>>(Wh_e, Whe_t, 64, 192);
    prep_wnmsg<<<(64 * 128 + 255) / 256, 256, 0, stream>>>(W_nmsg, Wnm_t);
    prep_wgru_n<<<(192 * 64 + 255) / 256, 256, 0, stream>>>(Wx_n, Wxn_t);
    prep_wgru_n<<<(192 * 64 + 255) / 256, 256, 0, stream>>>(Wh_n, Whn_t);
    prep_bias_n<<<1, 192, 0, stream>>>(b_nmsg, bx_n, bh_n, bnm_p, bxn_p, bhn_p);

    const int msgBlocks = (M + 127) / 128;   // 8 waves x 16 rows per block
    const int gruBlocks = (E + 127) / 128;

    // ---- edge (link) message passing, T = 4 -------------------------------
    unsigned short* lbIn = link_bf0;
    unsigned short* lbOut = link_bf1;
    for (int t = 0; t < 4; ++t) {
        (void)hipMemsetAsync(agg, 0, (size_t)E * 64 * 4, stream);
        msg_kernel<<<msgBlocks, 256, 0, stream>>>(lbIn, states_first, states_second,
                                                  Wm_t, b_msg, agg, M);
        gru_kernel<<<gruBlocks, 256, 0, stream>>>(agg, link_f, lbIn, lbOut,
                                                  Wxe_t, Whe_t, bx_e, bh_e, E);
        unsigned short* tmp = lbIn; lbIn = lbOut; lbOut = tmp;
    }

    // ---- node message passing, T = 4 --------------------------------------
    unsigned short* nbIn = node_bf0;
    unsigned short* nbOut = node_bf1;
    for (int t = 0; t < 4; ++t) {
        (void)hipMemsetAsync(agg, 0, (size_t)E * 64 * 4, stream);
        msg_kernel<<<msgBlocks, 256, 0, stream>>>(nbIn, states_first, states_second,
                                                  Wnm_t, bnm_p, agg, M);
        gru_kernel<<<gruBlocks, 256, 0, stream>>>(agg, node_f, nbIn, nbOut,
                                                  Wxn_t, Whn_t, bxn_p, bhn_p, E);
        unsigned short* tmp = nbIn; nbIn = nbOut; nbOut = tmp;
    }

    // ---- readout ----------------------------------------------------------
    (void)hipMemsetAsync(pooled, 0, (size_t)G * 128 * 4, stream);
    {
        long long total = (long long)E * 128;
        int blocks = (int)((total + 255) / 256);
        pool_kernel<<<blocks, 256, 0, stream>>>(link_f, node_f, states_gids, pooled, E);
    }
    mlp1_kernel<<<G, 256, 0, stream>>>(pooled, W1, b1, h1);
    mlp2_kernel<<<G, 256, 0, stream>>>(h1, W2, b2, h2);
    mlp3_kernel<<<(G + 255) / 256, 256, 0, stream>>>(h2, W3, b3, (float*)d_out, G);
}
